// HMGRec_46256797777931
// MI455X (gfx1250) — matrix operand order repaired and checked by reading, not compiled, not run on hardware
//
#include <hip/hip_runtime.h>
#include <math.h>

// ---------------------------------------------------------------------------
// HMGRec forward for MI455X (gfx1250, wave32).
//
// Roofline: head GEMMs = 38 GFLOP, 294 MB f32 output -> ~12.6us HBM floor.
// bf16 WMMA (v_wmma_f32_16x16x32_bf16) with K=128 exactly (component 0 of the
// tangent vectors is identically zero, so the k=0 column of Wd/emb drops out).
// A/B converted to bf16 once (~6MB, L2-resident); f32 accumulate.
// Head GEMM: per block, the 16KB A-panel (64 rows) is staged in LDS once and
// shared by all 8 waves (A L2 traffic /8); each wave does 4 M-tiles x 1
// N-tile with B fragments register-resident -> 8 WMMA per K-step, B-only
// global loads (4 b128), A frags via ds_load_b128 from padded LDS.
// ---------------------------------------------------------------------------

#define NDIM   128
#define HDIM   129
#define SEQ    100
#define BATCHN 64
#define MROWS  (BATCHN * SEQ)   // 6400
#define NPOI   10000
#define NCAT   500
#define NGEO   1000

#define LDSPITCH 136            // 128 bf16 + 8 pad (272B row stride, conflict-free)

typedef __attribute__((ext_vector_type(16))) __bf16 v16bf;
typedef __attribute__((ext_vector_type(8)))  __bf16 v8bf;
typedef __attribute__((ext_vector_type(8)))  float  v8f;

// ---------------------------------------------------------------------------
// K1: checkin[row, :] = projx(0.3*poi_tan + 0.1*cat_tan + 0.1*geo_tan + 0.5*user_tan)
//     (tangent vectors are just [0, emb[1:]], so only spatial parts combine)
// grid = MROWS blocks of 128 threads (thread t -> spatial dim 1+t)
// ---------------------------------------------------------------------------
__global__ void hmg_build_checkin(const int* __restrict__ poi_seqs,
                                  const int* __restrict__ cat_seqs,
                                  const int* __restrict__ geo_seqs,
                                  const int* __restrict__ user_list,
                                  const float* __restrict__ poi_emb,
                                  const float* __restrict__ cat_emb,
                                  const float* __restrict__ geo_emb,
                                  const float* __restrict__ user_emb,
                                  float* __restrict__ checkin) {
  int row = blockIdx.x;
  int b   = row / SEQ;
  int t   = threadIdx.x;  // 0..127
  int ip = poi_seqs[row], ic = cat_seqs[row], ig = geo_seqs[row], iu = user_list[b];
  float sp = 0.3f * poi_emb[(size_t)ip * HDIM + 1 + t]
           + 0.1f * cat_emb[(size_t)ic * HDIM + 1 + t]
           + 0.1f * geo_emb[(size_t)ig * HDIM + 1 + t]
           + 0.5f * user_emb[(size_t)iu * HDIM + 1 + t];
  float ss = sp * sp;
  #pragma unroll
  for (int o = 16; o > 0; o >>= 1) ss += __shfl_xor(ss, o, 32);
  __shared__ float part[4];
  if ((threadIdx.x & 31) == 0) part[threadIdx.x >> 5] = ss;
  __syncthreads();
  float tot = part[0] + part[1] + part[2] + part[3];
  checkin[(size_t)row * HDIM + 1 + t] = sp;
  if (t == 0) checkin[(size_t)row * HDIM] = sqrtf(1.0f + tot);
}

// ---------------------------------------------------------------------------
// K2: q / kf / v = checkin @ W^T + b   (kf has component 0 negated)
// grid = (MROWS, 3), block = 160 (thread d computes one output dim, d<129)
// ---------------------------------------------------------------------------
__global__ void hmg_qkv(const float* __restrict__ checkin,
                        const float* __restrict__ Wq, const float* __restrict__ bq,
                        const float* __restrict__ Wk, const float* __restrict__ bk,
                        const float* __restrict__ Wv, const float* __restrict__ bv,
                        float* __restrict__ q, float* __restrict__ kf,
                        float* __restrict__ v) {
  int row = blockIdx.x;
  int which = blockIdx.y;
  const float* W  = (which == 0) ? Wq : (which == 1) ? Wk : Wv;
  const float* bb = (which == 0) ? bq : (which == 1) ? bk : bv;
  float* outp     = (which == 0) ? q  : (which == 1) ? kf : v;
  __shared__ float x[HDIM];
  for (int i = threadIdx.x; i < HDIM; i += blockDim.x)
    x[i] = checkin[(size_t)row * HDIM + i];
  __syncthreads();
  int d = threadIdx.x;
  if (d < HDIM) {
    const float* wrow = W + (size_t)d * HDIM;
    float acc = bb[d];
    for (int k = 0; k < HDIM; ++k) acc = fmaf(x[k], wrow[k], acc);
    if (which == 1 && d == 0) acc = -acc;   // Lorentz flip folded into kf
    outp[(size_t)row * HDIM + d] = acc;
  }
}

// ---------------------------------------------------------------------------
// K3: Lorentz attention, one block (128 threads) per output row (b,s).
// ---------------------------------------------------------------------------
__global__ void hmg_attention(const float* __restrict__ q,
                              const float* __restrict__ kf,
                              const float* __restrict__ v,
                              const int* __restrict__ poi_seqs,
                              float* __restrict__ traj) {
  int row = blockIdx.x;
  int b   = row / SEQ;
  int tid = threadIdx.x;
  __shared__ float qs[HDIM];
  __shared__ float wsc[SEQ];
  __shared__ float redA[4], redB[4], redC[4];
  for (int i = tid; i < HDIM; i += 128) qs[i] = q[(size_t)row * HDIM + i];
  __syncthreads();

  const float scale = 0.08804509063f;  // 1/sqrt(129)
  for (int t = tid; t < SEQ; t += 128) {
    const float* krow = kf + (size_t)(b * SEQ + t) * HDIM;
    float inner = 0.0f;
    for (int k = 0; k < HDIM; ++k) inner = fmaf(qs[k], krow[k], inner);
    float sc = (2.0f + 2.0f * inner) * scale;
    if (poi_seqs[b * SEQ + t] == NPOI) sc = -1.0e9f;
    wsc[t] = sc;
  }
  __syncthreads();

  // softmax
  float mloc = -3.0e38f;
  for (int t = tid; t < SEQ; t += 128) mloc = fmaxf(mloc, wsc[t]);
  #pragma unroll
  for (int o = 16; o > 0; o >>= 1) mloc = fmaxf(mloc, __shfl_xor(mloc, o, 32));
  if ((tid & 31) == 0) redA[tid >> 5] = mloc;
  __syncthreads();
  float gmax = fmaxf(fmaxf(redA[0], redA[1]), fmaxf(redA[2], redA[3]));

  float sloc = 0.0f;
  for (int t = tid; t < SEQ; t += 128) {
    float e = __expf(wsc[t] - gmax);
    wsc[t] = e;
    sloc += e;
  }
  #pragma unroll
  for (int o = 16; o > 0; o >>= 1) sloc += __shfl_xor(sloc, o, 32);
  if ((tid & 31) == 0) redB[tid >> 5] = sloc;
  __syncthreads();
  float rinv = 1.0f / (redB[0] + redB[1] + redB[2] + redB[3]);

  // m = w @ v ; thread tid owns dim tid (+ dim 128 for tid==0)
  float a0 = 0.0f, a1 = 0.0f;
  for (int t = 0; t < SEQ; ++t) {
    float wt = wsc[t];
    const float* vrow = v + (size_t)(b * SEQ + t) * HDIM;
    a0 = fmaf(wt, vrow[tid], a0);
    if (tid == 0) a1 = fmaf(wt, vrow[128], a1);
  }
  a0 *= rinv; a1 *= rinv;

  // mm = -m0^2 + sum_{d>=1} m_d^2
  float c = (tid == 0) ? (-a0 * a0 + a1 * a1) : (a0 * a0);
  #pragma unroll
  for (int o = 16; o > 0; o >>= 1) c += __shfl_xor(c, o, 32);
  if ((tid & 31) == 0) redC[tid >> 5] = c;
  __syncthreads();
  float mm = redC[0] + redC[1] + redC[2] + redC[3];
  float denom = sqrtf(fmaxf(-mm, 1.0e-6f));
  float rden = 1.0f / denom;
  traj[(size_t)row * HDIM + tid] = a0 * rden;
  if (tid == 0) traj[(size_t)row * HDIM + 128] = a1 * rden;
}

// ---------------------------------------------------------------------------
// K4: rotate spatial pairs + tangent; emit bf16 A matrices (MROWS x 128).
// grid = (MROWS, 3), block = 64 (thread j handles pair j)
// ---------------------------------------------------------------------------
__global__ void hmg_rotate_tan(const float* __restrict__ traj,
                               const float* __restrict__ th_poi,
                               const float* __restrict__ th_cat,
                               const float* __restrict__ th_geo,
                               __bf16* __restrict__ t0,
                               __bf16* __restrict__ t1,
                               __bf16* __restrict__ t2) {
  int row = blockIdx.x;
  int h   = blockIdx.y;
  int j   = threadIdx.x;  // 0..63
  const float* th = (h == 0) ? th_poi : (h == 1) ? th_cat : th_geo;
  __bf16* outp    = (h == 0) ? t0 : (h == 1) ? t1 : t2;
  float s, c;
  __sincosf(th[j], &s, &c);
  float a  = traj[(size_t)row * HDIM + 1 + 2 * j];
  float bb = traj[(size_t)row * HDIM + 2 + 2 * j];
  outp[(size_t)row * NDIM + 2 * j]     = (__bf16)(a * c - bb * s);
  outp[(size_t)row * NDIM + 2 * j + 1] = (__bf16)(a * s + bb * c);
}

// ---------------------------------------------------------------------------
// K5: convert B matrix to bf16, dropping column 0 (dead: t[...,0]==0).
// src is (>=N) x 129 f32 row-major; dst is Npad x 128 bf16 (rows >= N zeroed).
// grid = Npad blocks of 128 threads
// ---------------------------------------------------------------------------
__global__ void hmg_convert_B(const float* __restrict__ src, int N,
                              __bf16* __restrict__ dst) {
  int n = blockIdx.x;
  int k = threadIdx.x;
  float vv = (n < N) ? src[(size_t)n * HDIM + 1 + k] : 0.0f;
  dst[(size_t)n * NDIM + k] = (__bf16)vv;
}

// ---------------------------------------------------------------------------
// K6: head GEMM.  out[m,n] = (A@Bw^T)[m,n] + bias[n] + exp(2 + 2*(A@Be^T)[m,n])
// A: MROWS x 128 bf16, Bw/Be: Npad x 128 bf16, out: MROWS x N f32.
// Block: stages the 64-row A-panel (16KB) in LDS once, shared by 8 waves.
// Each wave: 4 M-tiles x 1 N-tile; B frags register-resident across M-tiles.
// K=128 -> 4 K-steps, 8 WMMA per step, 32 WMMA per wave.
// Fragment layouts per CDNA5 ISA §7.12.2 (wave32).
// ---------------------------------------------------------------------------
__global__ void __launch_bounds__(256)
hmg_head_gemm(const __bf16* __restrict__ A,
              const __bf16* __restrict__ Bw,
              const __bf16* __restrict__ Be,
              const float* __restrict__ bias,
              float* __restrict__ out, int N, int Ntiles) {
  int lane = threadIdx.x & 31;
  int wave = threadIdx.x >> 5;
  int mg = blockIdx.y;                 // M-group: rows mg*64 .. mg*64+63
  int nt = blockIdx.x * 8 + wave;      // N-tile
  int half = lane >> 4;                // 0: lanes 0-15, 1: lanes 16-31
  int lr   = lane & 15;

  // ---- cooperative A-panel staging: 64 rows x 128 bf16, padded pitch ----
  __shared__ __bf16 As[64][LDSPITCH];
  {
    const __bf16* Abase = A + (size_t)mg * 64 * NDIM;
    #pragma unroll
    for (int c = 0; c < 4; ++c) {
      int chunk = threadIdx.x + c * 256;   // 0..1023, 16B each
      int row = chunk >> 4;                // /16 chunks per row
      int ck  = chunk & 15;                // 8-elem chunk within row
      *(v8bf*)(&As[row][ck * 8]) = *(const v8bf*)(Abase + (size_t)row * NDIM + ck * 8);
    }
  }
  __syncthreads();

  if (nt >= Ntiles) return;
  int n0 = nt * 16;

  const __bf16* BwRow = Bw + (size_t)(n0 + lr) * NDIM;   // B col n0+lr (N-major)
  const __bf16* BeRow = Be + (size_t)(n0 + lr) * NDIM;

  v8f accL[4] = {{}, {}, {}, {}};
  v8f accD[4] = {{}, {}, {}, {}};
  #pragma unroll
  for (int kb = 0; kb < NDIM; kb += 32) {
    // B frag: lanes<16 -> K = kb+{0..7,16..23}; lanes>=16 -> +8
    v8bf bw0 = *(const v8bf*)(BwRow + kb + half * 8);
    v8bf bw1 = *(const v8bf*)(BwRow + kb + 16 + half * 8);
    v16bf bw = __builtin_shufflevector(bw0, bw1,
        0, 1, 2, 3, 4, 5, 6, 7, 8, 9, 10, 11, 12, 13, 14, 15);
    v8bf be0 = *(const v8bf*)(BeRow + kb + half * 8);
    v8bf be1 = *(const v8bf*)(BeRow + kb + 16 + half * 8);
    v16bf be = __builtin_shufflevector(be0, be1,
        0, 1, 2, 3, 4, 5, 6, 7, 8, 9, 10, 11, 12, 13, 14, 15);

    #pragma unroll
    for (int j = 0; j < 4; ++j) {
      // A frag from LDS: lanes<16 -> K = kb+{0..7,16..23}; lanes>=16 -> +8
      const __bf16* Ar = &As[16 * j + lr][0];
      v8bf a_lo = *(const v8bf*)(Ar + kb + half * 8);
      v8bf a_hi = *(const v8bf*)(Ar + kb + 16 + half * 8);
      v16bf a = __builtin_shufflevector(a_lo, a_hi,
          0, 1, 2, 3, 4, 5, 6, 7, 8, 9, 10, 11, 12, 13, 14, 15);

      accL[j] = __builtin_amdgcn_wmma_f32_16x16x32_bf16(
          false, a, false, bw, (short)0, accL[j], false, false);
      accD[j] = __builtin_amdgcn_wmma_f32_16x16x32_bf16(
          false, a, false, be, (short)0, accD[j], false, false);
    }
  }

  // C/D layout: col = n0+lr (per lane), row = m_base + r + 8*half (per VGPR r)
  int n = n0 + lr;
  if (n < N) {
    float bn = bias[n];
    #pragma unroll
    for (int j = 0; j < 4; ++j) {
      float* orow = out + (size_t)(mg * 64 + j * 16 + 8 * half) * N + n;
      #pragma unroll
      for (int r = 0; r < 8; ++r) {
        orow[(size_t)r * N] = accL[j][r] + bn + __expf(2.0f + 2.0f * accD[j][r]);
      }
    }
  }
}

// ---------------------------------------------------------------------------
extern "C" void kernel_launch(void* const* d_in, const int* in_sizes, int n_in,
                              void* d_out, int out_size, void* d_ws, size_t ws_size,
                              hipStream_t stream) {
  const int*   poi_seqs  = (const int*)d_in[0];
  const int*   cat_seqs  = (const int*)d_in[1];
  const int*   geo_seqs  = (const int*)d_in[2];
  const int*   user_list = (const int*)d_in[3];
  const float* poi_emb   = (const float*)d_in[4];
  const float* cat_emb   = (const float*)d_in[5];
  const float* geo_emb   = (const float*)d_in[6];
  const float* user_emb  = (const float*)d_in[7];
  const float* th_poi    = (const float*)d_in[8];
  const float* th_cat    = (const float*)d_in[9];
  const float* th_geo    = (const float*)d_in[10];
  const float* Wq = (const float*)d_in[11]; const float* bq = (const float*)d_in[12];
  const float* Wk = (const float*)d_in[13]; const float* bk = (const float*)d_in[14];
  const float* Wv = (const float*)d_in[15]; const float* bv = (const float*)d_in[16];
  const float* Wd_poi = (const float*)d_in[17]; const float* bd_poi = (const float*)d_in[18];
  const float* Wd_cat = (const float*)d_in[19]; const float* bd_cat = (const float*)d_in[20];
  const float* Wd_geo = (const float*)d_in[21]; const float* bd_geo = (const float*)d_in[22];

  const int NcatP = 512;   // 500 -> 32 tiles
  const int NgeoP = 1008;  // 1000 -> 63 tiles

  // workspace carve-out (~27 MB total)
  char* ws = (char*)d_ws;
  size_t off = 0;
  auto carve = [&](size_t bytes) -> void* {
    void* p = ws + off;
    off = (off + bytes + 255) & ~(size_t)255;
    return p;
  };
  float* checkin = (float*)carve((size_t)MROWS * HDIM * 4);
  float* q       = (float*)carve((size_t)MROWS * HDIM * 4);
  float* kf      = (float*)carve((size_t)MROWS * HDIM * 4);
  float* v       = (float*)carve((size_t)MROWS * HDIM * 4);
  float* traj    = (float*)carve((size_t)MROWS * HDIM * 4);
  __bf16* tsp0   = (__bf16*)carve((size_t)MROWS * NDIM * 2);
  __bf16* tsp1   = (__bf16*)carve((size_t)MROWS * NDIM * 2);
  __bf16* tsp2   = (__bf16*)carve((size_t)MROWS * NDIM * 2);
  __bf16* BwPoi  = (__bf16*)carve((size_t)NPOI  * NDIM * 2);
  __bf16* BePoi  = (__bf16*)carve((size_t)NPOI  * NDIM * 2);
  __bf16* BwCat  = (__bf16*)carve((size_t)NcatP * NDIM * 2);
  __bf16* BeCat  = (__bf16*)carve((size_t)NcatP * NDIM * 2);
  __bf16* BwGeo  = (__bf16*)carve((size_t)NgeoP * NDIM * 2);
  __bf16* BeGeo  = (__bf16*)carve((size_t)NgeoP * NDIM * 2);

  float* out_poi = (float*)d_out;
  float* out_cat = out_poi + (size_t)MROWS * NPOI;
  float* out_geo = out_cat + (size_t)MROWS * NCAT;

  hmg_build_checkin<<<MROWS, 128, 0, stream>>>(
      poi_seqs, cat_seqs, geo_seqs, user_list,
      poi_emb, cat_emb, geo_emb, user_emb, checkin);

  hmg_qkv<<<dim3(MROWS, 3), 160, 0, stream>>>(
      checkin, Wq, bq, Wk, bk, Wv, bv, q, kf, v);

  hmg_attention<<<MROWS, 128, 0, stream>>>(q, kf, v, poi_seqs, traj);

  hmg_rotate_tan<<<dim3(MROWS, 3), 64, 0, stream>>>(
      traj, th_poi, th_cat, th_geo, tsp0, tsp1, tsp2);

  hmg_convert_B<<<NPOI,  128, 0, stream>>>(Wd_poi,  NPOI, BwPoi);
  hmg_convert_B<<<NPOI,  128, 0, stream>>>(poi_emb, NPOI, BePoi);  // emb[:-1]
  hmg_convert_B<<<NcatP, 128, 0, stream>>>(Wd_cat,  NCAT, BwCat);
  hmg_convert_B<<<NcatP, 128, 0, stream>>>(cat_emb, NCAT, BeCat);
  hmg_convert_B<<<NgeoP, 128, 0, stream>>>(Wd_geo,  NGEO, BwGeo);
  hmg_convert_B<<<NgeoP, 128, 0, stream>>>(geo_emb, NGEO, BeGeo);

  // head GEMMs: grid.y = 100 M-groups (4 M-tiles each); 8 waves = 8 N-tiles/block
  hmg_head_gemm<<<dim3(79, MROWS / 64), 256, 0, stream>>>(
      tsp0, BwPoi, BePoi, bd_poi, out_poi, NPOI, 625);
  hmg_head_gemm<<<dim3(4, MROWS / 64), 256, 0, stream>>>(
      tsp1, BwCat, BeCat, bd_cat, out_cat, NCAT, 32);
  hmg_head_gemm<<<dim3(8, MROWS / 64), 256, 0, stream>>>(
      tsp2, BwGeo, BeGeo, bd_geo, out_geo, NGEO, 63);
}
